// BahdanauAttention_23055384445095
// MI455X (gfx1250) — matrix operand order repaired and checked by reading, not compiled, not run on hardware
//
#include <hip/hip_runtime.h>
#include <hip/hip_bf16.h>

#define Bn 32
#define Sn 2048
#define Dn 1024
#define LDA 1032   // padded bf16 row stride for A panel in LDS (conflict-free b128)

typedef __bf16 bf16_t;
typedef bf16_t bf16x8 __attribute__((ext_vector_type(8)));
typedef bf16_t v16bf  __attribute__((ext_vector_type(16)));
typedef float  v8f    __attribute__((ext_vector_type(8)));
typedef float  f32x4  __attribute__((ext_vector_type(4)));
typedef unsigned short u16x4 __attribute__((ext_vector_type(4)));

union ABfrag { v16bf v; bf16x8 h[2]; };

static __device__ __forceinline__ unsigned short f2bf(float f) {
    unsigned int u = __float_as_uint(f);
    unsigned int r = (u + 0x7FFFu + ((u >> 16) & 1u)) >> 16;   // RNE
    return (unsigned short)r;
}

// ---------------- Kernel 0a: Wh (f32, row-major [e][d]) -> bf16 ----------------
__global__ void k0_cvt(const float* __restrict__ Wh, unsigned short* __restrict__ Whb) {
    int i = (blockIdx.x * 256 + threadIdx.x) * 4;               // 1M elems / 4
    f32x4 x = *(const f32x4*)(Wh + i);
    u16x4 o;
    o[0] = f2bf(x[0]); o[1] = f2bf(x[1]); o[2] = f2bf(x[2]); o[3] = f2bf(x[3]);
    *(u16x4*)(Whb + i) = o;
}

// ---------------- Kernel 0b: zero the score accumulator ----------------
__global__ void k0_zero(float* __restrict__ scr) {
    scr[blockIdx.x * 256 + threadIdx.x] = 0.f;
}

// ------- Kernel 0c: dfb[b][e] = sum_d dec[b,d]*Ws[e,d] + bh+bs(+bc) -------
__global__ void k0_decfeat(const float* __restrict__ dh, const float* __restrict__ Ws,
                           const float* __restrict__ bh, const float* __restrict__ bs,
                           const float* __restrict__ bc, const int* __restrict__ flag,
                           float* __restrict__ dfb) {
    int b = blockIdx.y;
    int e = blockIdx.x * 256 + threadIdx.x;
    __shared__ float dsh[Dn];
    for (int i = threadIdx.x; i < Dn; i += 256)
        dsh[i] = (i < 512) ? dh[b * 512 + i] : dh[Bn * 512 + b * 512 + (i - 512)];
    __syncthreads();
    const float* wr = Ws + (size_t)e * Dn;
    float acc = 0.f;
#pragma unroll 4
    for (int d = 0; d < Dn; d += 4)
        acc += dsh[d] * wr[d] + dsh[d+1] * wr[d+1] + dsh[d+2] * wr[d+2] + dsh[d+3] * wr[d+3];
    float bias = bh[e] + bs[e] + (flag[0] ? bc[e] : 0.f);
    dfb[(size_t)b * Dn + e] = acc + bias;
}

// ------------- Kernel 1: fused enc-GEMM(bf16 WMMA) + tanh + dot(v) -> score -------------
// Tile: M=64 rows (4 m-tiles) x N=256 cols (16 n-tiles) per workgroup, K=1024.
// grid = (n-split=4 [fastest -> L2 reuse of A panel], m-block=32, b=32)
__global__ void __launch_bounds__(256)
k1_score(const float* __restrict__ enc, const unsigned short* __restrict__ Whb,
         const float* __restrict__ dfb, const float* __restrict__ pcov,
         const float* __restrict__ Wc, const float* __restrict__ vvec,
         const int* __restrict__ flag, float* __restrict__ score_ws) {
    const int e0  = blockIdx.x * 256;      // n-split
    const int s0  = blockIdx.y * 64;       // m-block
    const int b   = blockIdx.z;
    const int tid = threadIdx.x;

    __shared__ unsigned short Ash[64 * LDA];   // 64 x 1024 bf16 panel, padded rows (132KB)
    __shared__ float covsh[64];
    __shared__ float scoresh[64];

    // Stage A panel: enc_output[b, s0..s0+63, :] f32 -> bf16 LDS
    const float* encB = enc + ((size_t)b * Sn + s0) * Dn;
    for (int i = tid; i < 16384; i += 256) {           // 16384 float4's
        int m  = i >> 8;                                // 256 float4 per row
        int d4 = (i & 255) * 4;
        f32x4 x = *(const f32x4*)(encB + m * Dn + d4);
        u16x4 o;
        o[0] = f2bf(x[0]); o[1] = f2bf(x[1]); o[2] = f2bf(x[2]); o[3] = f2bf(x[3]);
        *(u16x4*)&Ash[m * LDA + d4] = o;
    }
    if (tid < 64) { covsh[tid] = pcov[(size_t)b * Sn + s0 + tid]; scoresh[tid] = 0.f; }
    __syncthreads();

    const int wave = tid >> 5, lane = tid & 31;         // wave32
    const int grp  = lane >> 4, lrow = lane & 15;

    v8f acc[4][2];                                      // [m-tile][n-tile]
    const v8f vz = {0.f,0.f,0.f,0.f,0.f,0.f,0.f,0.f};
#pragma unroll
    for (int mt = 0; mt < 4; ++mt)
#pragma unroll
        for (int j = 0; j < 2; ++j) acc[mt][j] = vz;

    const unsigned short* brow0[2];
#pragma unroll
    for (int j = 0; j < 2; ++j) {
        int e = e0 + (wave * 2 + j) * 16 + lrow;        // B lane = column
        brow0[j] = Whb + (size_t)e * Dn + 8 * grp;
    }

    for (int k0 = 0; k0 < Dn; k0 += 32) {
        // A fragments: ISA 16-bit 16x32 layout, chunks at k0+8*grp and k0+16+8*grp
        ABfrag af[4];
#pragma unroll
        for (int mt = 0; mt < 4; ++mt) {
            int row = mt * 16 + lrow;
            af[mt].h[0] = *(const bf16x8*)&Ash[row * LDA + k0 + 8 * grp];
            af[mt].h[1] = *(const bf16x8*)&Ash[row * LDA + k0 + 16 + 8 * grp];
        }
        ABfrag bf[2];
#pragma unroll
        for (int j = 0; j < 2; ++j) {
            bf[j].h[0] = *(const bf16x8*)(brow0[j] + k0);
            bf[j].h[1] = *(const bf16x8*)(brow0[j] + k0 + 16);
        }
#pragma unroll
        for (int mt = 0; mt < 4; ++mt)
#pragma unroll
            for (int j = 0; j < 2; ++j)
                acc[mt][j] = __builtin_amdgcn_wmma_f32_16x16x32_bf16(
                                 false, af[mt].v, false, bf[j].v, (short)0,
                                 acc[mt][j], false, false);
    }

    // Fused epilogue: +dec_features +coverage*Wc, tanh, * v[e], partial sum over e
    const int cf = flag[0];
    float sc[4][8];
#pragma unroll
    for (int mt = 0; mt < 4; ++mt)
#pragma unroll
        for (int r = 0; r < 8; ++r) sc[mt][r] = 0.f;

#pragma unroll
    for (int j = 0; j < 2; ++j) {
        int e = e0 + (wave * 2 + j) * 16 + lrow;        // D-matrix column for this lane
        float fe = dfb[(size_t)b * Dn + e];
        float wc = Wc[e];
        float ve = vvec[e];
#pragma unroll
        for (int mt = 0; mt < 4; ++mt)
#pragma unroll
            for (int r = 0; r < 8; ++r) {               // row m = mt*16 + r + 8*grp
                float x = acc[mt][j][r] + fe;
                if (cf) x += covsh[mt * 16 + r + 8 * grp] * wc;
                sc[mt][r] += tanhf(x) * ve;
            }
    }
#pragma unroll
    for (int mt = 0; mt < 4; ++mt)
#pragma unroll
        for (int r = 0; r < 8; ++r)
            atomicAdd(&scoresh[mt * 16 + r + 8 * grp], sc[mt][r]);
    __syncthreads();
    if (tid < 64)
        atomicAdd(&score_ws[(size_t)b * Sn + s0 + tid], scoresh[tid]);  // across n-splits
}

// ------------- Kernel 2: masked softmax + renorm + coverage; zero context -------------
__global__ void k2_softmax(const float* __restrict__ score, const float* __restrict__ mask,
                           const float* __restrict__ pcov, const int* __restrict__ flag,
                           const float* __restrict__ bv,
                           float* __restrict__ out, float* __restrict__ w_ws) {
    int b = blockIdx.x, tid = threadIdx.x;
    float* ctx = out;                       // B*D
    float* aw  = out + Bn * Dn;             // B*S
    float* cov = out + Bn * Dn + Bn * Sn;   // B*S
    __shared__ float red[256];

    const float* sc = score + (size_t)b * Sn;
    float bvv = bv[0];
    float vals[8];
    float mx = -3.4e38f;
#pragma unroll
    for (int j = 0; j < 8; ++j) { vals[j] = sc[j * 256 + tid] + bvv; mx = fmaxf(mx, vals[j]); }
    red[tid] = mx; __syncthreads();
    for (int off = 128; off; off >>= 1) {
        if (tid < off) red[tid] = fmaxf(red[tid], red[tid + off]);
        __syncthreads();
    }
    mx = red[0]; __syncthreads();

    float lsum = 0.f;
#pragma unroll
    for (int j = 0; j < 8; ++j) {
        int s = j * 256 + tid;
        float e = __expf(vals[j] - mx) * mask[(size_t)b * Sn + s];  // fold mask pre-sum
        vals[j] = e; lsum += e;
    }
    red[tid] = lsum; __syncthreads();
    for (int off = 128; off; off >>= 1) {
        if (tid < off) red[tid] += red[tid + off];
        __syncthreads();
    }
    float inv = 1.0f / red[0];
    int cf = flag[0];
#pragma unroll
    for (int j = 0; j < 8; ++j) {
        int s = j * 256 + tid;
        float w = vals[j] * inv;
        aw[(size_t)b * Sn + s]   = w;
        w_ws[(size_t)b * Sn + s] = w;
        if (cf) cov[(size_t)b * Sn + s] = pcov[(size_t)b * Sn + s] + w;
    }
    for (int d = tid; d < Dn; d += 256) ctx[(size_t)b * Dn + d] = 0.f;  // for K3 atomics
}

// ------------- Kernel 3: context[b,d] = sum_s w[b,s] * enc[b,s,d] -------------
__global__ void k3_context(const float* __restrict__ enc, const float* __restrict__ w_ws,
                           float* __restrict__ out) {
    int tid = threadIdx.x;
    int d  = blockIdx.x * 256 + tid;
    int b  = blockIdx.y;
    int s0 = blockIdx.z * 256;
    __shared__ float wsh[256];
    wsh[tid] = w_ws[(size_t)b * Sn + s0 + tid];
    __syncthreads();
    const float* ep = enc + ((size_t)b * Sn + s0) * Dn + d;
    float acc = 0.f;
#pragma unroll 4
    for (int s = 0; s < 256; ++s) acc += wsh[s] * ep[(size_t)s * Dn];
    atomicAdd(&out[(size_t)b * Dn + d], acc);
}

extern "C" void kernel_launch(void* const* d_in, const int* in_sizes, int n_in,
                              void* d_out, int out_size, void* d_ws, size_t ws_size,
                              hipStream_t stream) {
    const float* dh   = (const float*)d_in[0];
    const float* enc  = (const float*)d_in[1];
    const float* mask = (const float*)d_in[2];
    const float* pcov = (const float*)d_in[3];
    const float* Wh   = (const float*)d_in[4];
    const float* bh   = (const float*)d_in[5];
    const float* Ws   = (const float*)d_in[6];
    const float* bs   = (const float*)d_in[7];
    const float* Wc   = (const float*)d_in[8];
    const float* bc   = (const float*)d_in[9];
    const float* vv   = (const float*)d_in[10];
    const float* bv   = (const float*)d_in[11];
    const int*   flag = (const int*)d_in[12];
    float* out = (float*)d_out;

    // Workspace layout (~2.75 MB total)
    char* ws = (char*)d_ws;
    unsigned short* Whb = (unsigned short*)ws;                       // 2 MB bf16 Wh
    float* dfb  = (float*)(ws + 2097152);                            // 128 KB
    float* scr  = (float*)(ws + 2097152 + 131072);                   // 256 KB scores
    float* wbuf = (float*)(ws + 2097152 + 131072 + 262144);          // 256 KB weights

    k0_cvt    <<<dim3(1024), 256, 0, stream>>>(Wh, Whb);
    k0_zero   <<<dim3(Bn * Sn / 256), 256, 0, stream>>>(scr);
    k0_decfeat<<<dim3(4, Bn), 256, 0, stream>>>(dh, Ws, bh, bs, bc, flag, dfb);
    k1_score  <<<dim3(4, Sn / 64, Bn), 256, 0, stream>>>(enc, Whb, dfb, pcov, Wc, vv, flag, scr);
    k2_softmax<<<dim3(Bn), 256, 0, stream>>>(scr, mask, pcov, flag, bv, out, wbuf);
    k3_context<<<dim3(Dn / 256, Bn, Sn / 256), 256, 0, stream>>>(enc, wbuf, out);
}
